// PatchCoreDINOv2_18674517803021
// MI455X (gfx1250) — compile-verified, matrix operand order checked by reading, not run on hardware
//
#include <hip/hip_runtime.h>
#include <stdint.h>

typedef __attribute__((ext_vector_type(16))) __bf16 v16bf;
typedef __attribute__((ext_vector_type(8)))  float  v8f;

union FragBF {
  v16bf v;
  uint32_t u[8];
};

#define D_DIM 384

__device__ __forceinline__ uint16_t f32_to_bf16(float f) {
  uint32_t u = __float_as_uint(f);
  uint32_t r = 0x7FFFu + ((u >> 16) & 1u);   // round-to-nearest-even
  return (uint16_t)((u + r) >> 16);
}

// ---------------------------------------------------------------------------
// Kernel 1: L2-normalize patch rows -> bf16 A matrix + p_sq, init row_min=+inf
// grid = rows (1024), block = 128
// ---------------------------------------------------------------------------
__global__ void k_normalize(const float* __restrict__ patches,
                            uint16_t* __restrict__ Abf,
                            float* __restrict__ p_sq,
                            uint32_t* __restrict__ row_min) {
  __shared__ float red[128];
  const int r = blockIdx.x;
  const int t = threadIdx.x;
  const float* src = patches + (size_t)r * D_DIM;

  float s = 0.f;
  for (int d = t; d < D_DIM; d += 128) { float x = src[d]; s += x * x; }
  red[t] = s; __syncthreads();
  for (int w = 64; w > 0; w >>= 1) {
    if (t < w) red[t] += red[t + w];
    __syncthreads();
  }
  const float tot = red[0];
  const float inv = 1.0f / (sqrtf(tot) + 1e-12f);
  for (int d = t; d < D_DIM; d += 128)
    Abf[(size_t)r * D_DIM + d] = f32_to_bf16(src[d] * inv);
  if (t == 0) {
    p_sq[r]    = tot * inv * inv;    // ||p_normalized||^2, as reference computes it
    row_min[r] = 0x7f800000u;        // +inf bits (re-init every launch)
  }
}

// ---------------------------------------------------------------------------
// Kernel 2: main WMMA distance kernel.
// grid = (ceil(N/64), ceil(rows/256)), block = 256 (8 waves).
// Block stages a 64-row bank tile (fp32 -> bf16) into LDS with per-row sum of
// squares. Each wave owns a 32-row x 64-col output tile: each B fragment feeds
// two WMMAs, and all four B fragments of a k-step are preloaded (double
// buffering in registers) so the 8 ds_load_b128 can clause ahead of the
// 8-WMMA burst instead of serializing on s_wait_dscnt.
// ---------------------------------------------------------------------------
__global__ void __launch_bounds__(256)
k_patch_min(const uint16_t* __restrict__ Abf,
            const float* __restrict__ p_sq,
            const float* __restrict__ mb_local,
            uint32_t* __restrict__ row_min,
            int rows, int Nn) {
  __shared__ uint32_t Bs32[64 * 192];   // 64 rows x 384 bf16 = 49152 B
  __shared__ float    sqs[64];

  const int tid = threadIdx.x;
  const int n0  = blockIdx.x * 64;

  if (tid < 64) sqs[tid] = 0.f;
  __syncthreads();

  // ---- stage bank tile: 4 threads per row, 24 float4 each ----
  {
    const int row = tid >> 2;          // 0..63
    const int q   = tid & 3;           // quarter of the row
    const int n   = n0 + row;
    const float4* src = (const float4*)mb_local;  // 96 float4 per row
    float ss = 0.f;
    #pragma unroll 4
    for (int i = 0; i < 24; ++i) {
      const int f4 = q * 24 + i;       // 0..95
      float4 vv;
      if (n < Nn) vv = src[(size_t)n * 96 + f4];
      else { vv.x = vv.y = vv.z = vv.w = 0.f; }
      ss += vv.x * vv.x + vv.y * vv.y + vv.z * vv.z + vv.w * vv.w;
      const uint32_t lo = (uint32_t)f32_to_bf16(vv.x) | ((uint32_t)f32_to_bf16(vv.y) << 16);
      const uint32_t hi = (uint32_t)f32_to_bf16(vv.z) | ((uint32_t)f32_to_bf16(vv.w) << 16);
      const int di = row * 192 + f4 * 2;
      Bs32[di]     = lo;
      Bs32[di + 1] = hi;
    }
    if (n < Nn) atomicAdd(&sqs[row], ss);   // ds_add_f32; matches fp32 mbl_sq
  }
  __syncthreads();

  const int lane = tid & 31;
  const int wave = tid >> 5;
  const int r0   = blockIdx.y * 256 + wave * 32;   // 32 rows per wave
  if (r0 >= rows) return;   // wave-uniform; no barriers past this point

  const int lr  = lane & 15;   // M (for A) / N (for B) within fragment
  const int kh  = lane >> 4;   // K-half selector
  const int kh4 = kh * 4;

  const v8f vzero = {0.f, 0.f, 0.f, 0.f, 0.f, 0.f, 0.f, 0.f};
  v8f acc[2][4];
  #pragma unroll
  for (int t2 = 0; t2 < 2; ++t2)
    #pragma unroll
    for (int j = 0; j < 4; ++j) acc[t2][j] = vzero;

  // Precompute pointers so the k-loop is pure pointer-bump + immediate offsets.
  const uint32_t* A32 = (const uint32_t*)Abf;
  const uint32_t* aP0 = A32 + ((size_t)(r0 + lr)      * D_DIM >> 1);
  const uint32_t* aP1 = A32 + ((size_t)(r0 + 16 + lr) * D_DIM >> 1);
  int bOff = lr * 192;   // dword index into Bs32; +16 dwords per k-step

  for (int kt = 0; kt < 12; ++kt) {
    FragBF a0, a1;
    #pragma unroll
    for (int v = 0; v < 8; ++v) {
      // ISA 16-bit A 16x32 layout: K = (v/4)*16 + kh*8 + (v%4)*2  (dword units: >>1)
      const int kOff = (v >> 2) * 8 + kh4 + (v & 3);
      a0.u[v] = aP0[kOff];
      a1.u[v] = aP1[kOff];
    }
    FragBF b[4];
    #pragma unroll
    for (int j = 0; j < 4; ++j) {
      #pragma unroll
      for (int v = 0; v < 8; ++v) {
        const int kOff = (v >> 2) * 8 + kh4 + (v & 3);
        b[j].u[v] = Bs32[bOff + j * (16 * 192) + kOff];
      }
    }
    #pragma unroll
    for (int j = 0; j < 4; ++j) {
      acc[0][j] = __builtin_amdgcn_wmma_f32_16x16x32_bf16(
          false, a0.v, false, b[j].v, (short)0, acc[0][j], false, false);
      acc[1][j] = __builtin_amdgcn_wmma_f32_16x16x32_bf16(
          false, a1.v, false, b[j].v, (short)0, acc[1][j], false, false);
    }
    aP0 += 16; aP1 += 16; bOff += 16;
  }

  // ---- epilogue: d2 = p_sq + mbl_sq - 2*dot, clamp, min over n, atomic min ----
  float sqv[4];
  #pragma unroll
  for (int j = 0; j < 4; ++j) {
    const int n = n0 + j * 16 + lr;
    sqv[j] = (n < Nn) ? sqs[j * 16 + lr] : __builtin_inff();
  }
  #pragma unroll
  for (int t2 = 0; t2 < 2; ++t2) {
    #pragma unroll
    for (int i = 0; i < 8; ++i) {
      const int rrow = r0 + t2 * 16 + kh * 8 + i;  // C/D layout: M = (lane/16)*8 + vgpr
      const float pv = p_sq[rrow];
      float m = __builtin_inff();
      #pragma unroll
      for (int j = 0; j < 4; ++j) {
        float d2 = pv + sqv[j] - 2.0f * acc[t2][j][i];
        d2 = fmaxf(d2, 0.f);              // clamp commutes with min/max
        m = fminf(m, d2);
      }
      // butterfly min across the 16 lanes sharing this row (masks < 16 stay in half)
      m = fminf(m, __shfl_xor(m, 1, 32));
      m = fminf(m, __shfl_xor(m, 2, 32));
      m = fminf(m, __shfl_xor(m, 4, 32));
      m = fminf(m, __shfl_xor(m, 8, 32));
      if (lr == 0)   // lanes 0 and 16 publish their half's row
        atomicMin(&row_min[rrow], __float_as_uint(m));  // non-negative: bit order == float order
    }
  }
}

// ---------------------------------------------------------------------------
// Kernel 3: finalize. max over patches + sqrt, global branch, blend. 1 block.
// ---------------------------------------------------------------------------
__global__ void k_finalize(const uint32_t* __restrict__ row_min,
                           const float* __restrict__ globals_x,
                           const float* __restrict__ mb_global,
                           float* __restrict__ out,
                           int Bb, int P, int Mm) {
  __shared__ float red[256];
  __shared__ float gbuf[D_DIM];
  __shared__ float lmax_s[8];
  const int t = threadIdx.x;

  // local part: max over patches of per-row min d2
  for (int b = 0; b < Bb; ++b) {
    float vmax = 0.f;   // d2 >= 0
    for (int p = t; p < P; p += 256)
      vmax = fmaxf(vmax, __uint_as_float(row_min[b * P + p]));
    red[t] = vmax; __syncthreads();
    for (int w = 128; w > 0; w >>= 1) {
      if (t < w) red[t] = fmaxf(red[t], red[t + w]);
      __syncthreads();
    }
    if (t == 0) lmax_s[b] = sqrtf(red[0]);
    __syncthreads();
  }

  // global part: min over mb_global of ||g_norm - mb||
  for (int b = 0; b < Bb; ++b) {
    const float* g = globals_x + (size_t)b * D_DIM;
    float s = 0.f;
    for (int d = t; d < D_DIM; d += 256) { float x = g[d]; s += x * x; }
    red[t] = s; __syncthreads();
    for (int w = 128; w > 0; w >>= 1) {
      if (t < w) red[t] += red[t + w];
      __syncthreads();
    }
    const float tot = red[0];
    __syncthreads();
    const float inv = 1.0f / (sqrtf(tot) + 1e-12f);
    const float gsq = tot * inv * inv;
    for (int d = t; d < D_DIM; d += 256) gbuf[d] = g[d] * inv;
    __syncthreads();

    float mn = __builtin_inff();
    for (int mi = t; mi < Mm; mi += 256) {
      const float* wrow = mb_global + (size_t)mi * D_DIM;
      float dot = 0.f, msq = 0.f;
      for (int d = 0; d < D_DIM; ++d) {
        const float x = wrow[d];
        dot += gbuf[d] * x;
        msq += x * x;
      }
      const float d2 = gsq + msq - 2.f * dot;
      mn = fminf(mn, fmaxf(d2, 0.f));
    }
    red[t] = mn; __syncthreads();
    for (int w = 128; w > 0; w >>= 1) {
      if (t < w) red[t] = fminf(red[t], red[t + w]);
      __syncthreads();
    }
    if (t == 0) out[b] = 0.7f * lmax_s[b] + (1.0f - 0.7f) * sqrtf(red[0]);
    __syncthreads();
  }
}

// ---------------------------------------------------------------------------
extern "C" void kernel_launch(void* const* d_in, const int* in_sizes, int n_in,
                              void* d_out, int out_size, void* d_ws, size_t ws_size,
                              hipStream_t stream) {
  (void)n_in; (void)out_size; (void)ws_size;
  const float* patches   = (const float*)d_in[0];
  const float* globals_x = (const float*)d_in[1];
  const float* mb_local  = (const float*)d_in[2];
  const float* mb_global = (const float*)d_in[3];
  float* out = (float*)d_out;

  const int D    = D_DIM;
  const int rows = in_sizes[0] / D;   // B*P = 1024
  const int Bb   = in_sizes[1] / D;   // 4
  const int Nn   = in_sizes[2] / D;   // 100000
  const int Mm   = in_sizes[3] / D;   // 2000
  const int P    = rows / Bb;         // 256

  // workspace layout
  uint16_t* Abf = (uint16_t*)d_ws;
  size_t off = (size_t)rows * D * sizeof(uint16_t);
  float* p_sq = (float*)((char*)d_ws + off);
  off += (size_t)rows * sizeof(float);
  uint32_t* row_min = (uint32_t*)((char*)d_ws + off);

  k_normalize<<<rows, 128, 0, stream>>>(patches, Abf, p_sq, row_min);

  dim3 grid((Nn + 63) / 64, (rows + 255) / 256);
  k_patch_min<<<grid, 256, 0, stream>>>(Abf, p_sq, mb_local, row_min, rows, Nn);

  k_finalize<<<1, 256, 0, stream>>>(row_min, globals_x, mb_global, out, Bb, P, Mm);
}
